// LSTMBlock_43654047596748
// MI455X (gfx1250) — matrix operand order repaired and checked by reading, not compiled, not run on hardware
//
#include <hip/hip_runtime.h>
#include <hip/hip_bf16.h>

typedef __bf16 bhalf;
typedef __attribute__((ext_vector_type(16))) __bf16 v16bf;
typedef __attribute__((ext_vector_type(8)))  __bf16 v8bf;
typedef __attribute__((ext_vector_type(8)))  float  v8f;

#define TB 2            // batch
#define TT 8            // timesteps
#define HH 14
#define WW 14
#define CC 512          // input channels (same for all 4 convs here)
#define FF 512          // LSTM filters
#define N4 2048         // 4*FF gate channels
#define HWP (HH*WW)     // 196
#define KT (9*CC)       // 4608 im2col K

__device__ __forceinline__ float hsig(float v) {
    return fminf(fmaxf(0.2f * v + 0.5f, 0.f), 1.f);
}

// ---------------------------------------------------------------------------
// Weight convert + transpose: src f32 [K=9*512][N=2048] -> dst bf16 [N][K].
// ---------------------------------------------------------------------------
__global__ void k_wconvert(const float* __restrict__ src, bhalf* __restrict__ dst) {
    long long i = (long long)blockIdx.x * blockDim.x + threadIdx.x;
    const long long total = (long long)N4 * KT;
    if (i >= total) return;
    int n = (int)(i / KT);
    int k = (int)(i - (long long)n * KT);
    dst[i] = (bhalf)src[(long long)k * N4 + n];
}

// Activation convert f32 -> bf16 (for GEMM A operands)
__global__ void k_acvt(const float* __restrict__ src, bhalf* __restrict__ dst, int n) {
    int i = blockIdx.x * blockDim.x + threadIdx.x;
    if (i < n) dst[i] = (bhalf)src[i];
}

__global__ void k_zero(float* __restrict__ p, int n) {
    int i = blockIdx.x * blockDim.x + threadIdx.x;
    if (i < n) p[i] = 0.f;
}

__global__ void k_zero_bf(bhalf* __restrict__ p, int n) {
    int i = blockIdx.x * blockDim.x + threadIdx.x;
    if (i < n) p[i] = (bhalf)0.f;
}

// ---------------------------------------------------------------------------
// Implicit-GEMM 3x3 SAME conv, bf16 WMMA, f32 accumulate.
//   A: bf16 activations [nb,14,14,512] gathered im2col-style
//   B: pre-transposed bf16 weights [N=2048][K=4608]
//   out[m][n] = (bias?bias[n]:0) + (addend?addend[m][n]:0) + A*B
// Block = 64 threads (2 waves). Each wave computes a 32(M) x 64(N) tile:
// two A fragments (row groups +0..15, +16..31), 4 shared B fragments,
// 8 f32 accumulators. Tap loop hoists all im2col boundary logic: invalid
// taps read from a zeroed 1KB pad buffer, so the inner loop is branch-free.
// ---------------------------------------------------------------------------
__global__ __launch_bounds__(64)
void k_conv_gemm(const bhalf* __restrict__ inp, int inBS,
                 const bhalf* __restrict__ zpad,
                 const bhalf* __restrict__ Wt,
                 const float* __restrict__ bias,
                 const float* __restrict__ addend, int adBS,
                 float* __restrict__ out, int outBS,
                 int M)
{
    const int lane = threadIdx.x & 31;
    const int wave = threadIdx.x >> 5;
    const int lm = lane & 15;       // position within half-wave
    const int lh = lane >> 4;       // which half-wave
    const int mw = blockIdx.x * 64 + wave * 32;
    const int n0 = blockIdx.y * 64;

    // Per-row-group im2col decomposition (loop invariant)
    const bhalf* abase[2];
    int ryg[2], rxg[2];
    bool rv[2];
#pragma unroll
    for (int g = 0; g < 2; ++g) {
        int row = mw + g * 16 + lm;
        bool v = row < M;
        int b = v ? row / HWP : 0;
        int p = v ? row - b * HWP : 0;
        ryg[g] = p / WW;
        rxg[g] = p - ryg[g] * WW;
        rv[g]  = v;
        abase[g] = inp + (size_t)b * inBS;
    }

    v8f acc[2][4] = {};
    // B base: column n0+lm, K-offset lh*16 (per-j offset j*16 rows of K)
    const bhalf* wp = Wt + (size_t)(n0 + lm) * KT + lh * 16;

    for (int tap = 0; tap < 9; ++tap) {
        const int r = tap / 3;
        const int s = tap - r * 3;
        const bhalf* ap[2];
#pragma unroll
        for (int g = 0; g < 2; ++g) {
            int yy = ryg[g] + r - 1;
            int xx = rxg[g] + s - 1;
            bool ok = rv[g] && (unsigned)yy < (unsigned)HH && (unsigned)xx < (unsigned)WW;
            ap[g] = ok ? (abase[g] + ((size_t)(yy * WW + xx) << 9) + lh * 16)
                       : (zpad + lh * 16);
        }
        const int kt0 = tap * CC;
        for (int c = 0; c < CC; c += 32) {
            // A fragments (contiguous k order, same as B): lane<16 -> K {0..15}, lane>=16 -> {16..31}
            v16bf a[2];
#pragma unroll
            for (int g = 0; g < 2; ++g) {
                v8bf c0 = *reinterpret_cast<const v8bf*>(ap[g] + c);
                v8bf c1 = *reinterpret_cast<const v8bf*>(ap[g] + c + 8);
#pragma unroll
                for (int i = 0; i < 8; i++) { a[g][i] = c0[i]; a[g][i + 8] = c1[i]; }
            }
#pragma unroll
            for (int j = 0; j < 4; j++) {
                v16bf b = *reinterpret_cast<const v16bf*>(wp + (size_t)j * 16 * KT + kt0 + c);
                acc[0][j] = __builtin_amdgcn_wmma_f32_16x16x32_bf16(
                    false, a[0], false, b, (short)0, acc[0][j], false, false);
                acc[1][j] = __builtin_amdgcn_wmma_f32_16x16x32_bf16(
                    false, a[1], false, b, (short)0, acc[1][j], false, false);
            }
        }
    }

    // C/D layout: VGPR i -> row mw + g*16 + lh*8 + i, col n0 + j*16 + lm
#pragma unroll
    for (int g = 0; g < 2; ++g) {
#pragma unroll
        for (int i = 0; i < 8; i++) {
            const int rr = mw + g * 16 + lh * 8 + i;
            if (rr >= M) continue;
            const int b2 = rr / HWP;
            const int p2 = rr - b2 * HWP;
            const size_t ob = (size_t)b2 * outBS + (size_t)p2 * N4;
            const size_t ab = (size_t)b2 * adBS  + (size_t)p2 * N4;
#pragma unroll
            for (int j = 0; j < 4; j++) {
                const int col = n0 + j * 16 + lm;
                float v = acc[g][j][i];
                if (bias)   v += bias[col];
                if (addend) v += addend[ab + col];
                out[ob + col] = v;
            }
        }
    }
}

// ---------------------------------------------------------------------------
// LSTM gate fusion (Keras gate order i,f,g,o; hard_sigmoid rec., relu act.)
// z: [TB][HWP][2048]; c: [TB][HWP][512] f32; h out as bf16 (next GEMM's A);
// yout: [TB][TT][HWP][512] f32 (write slot t)
// ---------------------------------------------------------------------------
__global__ void k_lstm_gate(const float* __restrict__ z, float* __restrict__ c,
                            bhalf* __restrict__ hbf, float* __restrict__ yout, int t)
{
    int i = blockIdx.x * blockDim.x + threadIdx.x;
    const int total = TB * HWP * FF;
    if (i >= total) return;
    int f = i & (FF - 1);
    int p = (i >> 9) % HWP;
    int b = i / (FF * HWP);
    const float* zp = z + ((size_t)b * HWP + p) * N4;
    float zi = zp[f], zf = zp[FF + f], zg = zp[2 * FF + f], zo = zp[3 * FF + f];
    float cn = hsig(zf) * c[i] + hsig(zi) * fmaxf(zg, 0.f);
    float hn = hsig(zo) * fmaxf(cn, 0.f);
    c[i]   = cn;
    hbf[i] = (bhalf)hn;
    yout[(((size_t)b * TT + t) * HWP + p) * FF + f] = hn;
}

// In-place batchnorm over [*, FF], eps = 1e-3 (Keras default)
__global__ void k_bn(float* __restrict__ y, const float* __restrict__ g,
                     const float* __restrict__ be, const float* __restrict__ mn,
                     const float* __restrict__ vr, int n)
{
    int i = blockIdx.x * blockDim.x + threadIdx.x;
    if (i >= n) return;
    int f = i & (FF - 1);
    y[i] = (y[i] - mn[f]) * rsqrtf(vr[f] + 1e-3f) * g[f] + be[f];
}

// Global average pool over HW fused with BN (affine BN commutes with the mean)
__global__ void k_bnpool(const float* __restrict__ y, const float* __restrict__ g,
                         const float* __restrict__ be, const float* __restrict__ mn,
                         const float* __restrict__ vr, float* __restrict__ out)
{
    int i = blockIdx.x * blockDim.x + threadIdx.x;
    const int total = TB * TT * FF;
    if (i >= total) return;
    int f  = i & (FF - 1);
    int bt = i >> 9;
    const float* p = y + (size_t)bt * HWP * FF + f;
    float s = 0.f;
    for (int q = 0; q < HWP; q++) s += p[(size_t)q * FF];
    float v = s * (1.f / (float)HWP);
    out[i] = (v - mn[f]) * rsqrtf(vr[f] + 1e-3f) * g[f] + be[f];
}

// ---------------------------------------------------------------------------
extern "C" void kernel_launch(void* const* d_in, const int* in_sizes, int n_in,
                              void* d_out, int out_size, void* d_ws, size_t ws_size,
                              hipStream_t stream)
{
    (void)in_sizes; (void)n_in; (void)out_size; (void)ws_size;
    const float* x   = (const float*)d_in[0];
    const float* k1  = (const float*)d_in[1];
    const float* rk1 = (const float*)d_in[2];
    const float* b1  = (const float*)d_in[3];
    const float* g1  = (const float*)d_in[4];
    const float* be1 = (const float*)d_in[5];
    const float* m1  = (const float*)d_in[6];
    const float* v1  = (const float*)d_in[7];
    const float* k2  = (const float*)d_in[8];
    const float* rk2 = (const float*)d_in[9];
    const float* b2  = (const float*)d_in[10];
    const float* g2  = (const float*)d_in[11];
    const float* be2 = (const float*)d_in[12];
    const float* m2  = (const float*)d_in[13];
    const float* v2  = (const float*)d_in[14];
    float* out = (float*)d_out;

    // Workspace carve (≈84 MB total)
    char* ws = (char*)d_ws;
    size_t off = 0;
    auto carve = [&](size_t bytes) {
        void* p = ws + off;
        off = (off + bytes + 255) & ~(size_t)255;
        return p;
    };
    bhalf* wtA  = (bhalf*)carve((size_t)N4 * KT * 2);            // input-conv weights bf16 [N][K]
    bhalf* wtB  = (bhalf*)carve((size_t)N4 * KT * 2);            // recurrent weights
    float* xc   = (float*)carve((size_t)TB * TT * HWP * N4 * 4); // precomputed input convs
    float* zb   = (float*)carve((size_t)TB * HWP * N4 * 4);      // per-step gate pre-activations
    bhalf* abf  = (bhalf*)carve((size_t)TB * TT * HWP * CC * 2); // bf16 input activations
    bhalf* hbf  = (bhalf*)carve((size_t)TB * HWP * FF * 2);      // bf16 h state
    float* cb   = (float*)carve((size_t)TB * HWP * FF * 4);      // c state
    float* y1   = (float*)carve((size_t)TB * TT * HWP * FF * 4); // layer-1 output
    float* y2   = (float*)carve((size_t)TB * TT * HWP * FF * 4); // layer-2 output
    bhalf* zpad = (bhalf*)carve((size_t)CC * 2);                 // zero pad row (SAME padding)

    const long long wcvN = (long long)N4 * KT;
    dim3 wcvGrid((unsigned)((wcvN + 255) / 256));
    const int stateN = TB * HWP * FF;
    dim3 elemGrid((stateN + 255) / 256);
    const int actN = TB * TT * HWP * CC;
    dim3 actGrid((actN + 255) / 256);
    dim3 gemmBlk(64);
    dim3 xGrid((TB * TT * HWP + 63) / 64, N4 / 64);   // 49 x 32
    dim3 rGrid((TB * HWP + 63) / 64, N4 / 64);        //  7 x 32
    const int bnN = TB * TT * HWP * FF;
    dim3 bnGrid((bnN + 255) / 256);

    // -------- Layer 1 --------
    k_wconvert<<<wcvGrid, 256, 0, stream>>>(k1,  wtA);
    k_wconvert<<<wcvGrid, 256, 0, stream>>>(rk1, wtB);
    k_acvt<<<actGrid, 256, 0, stream>>>(x, abf, actN);
    k_zero<<<elemGrid, 256, 0, stream>>>(cb, stateN);
    k_zero_bf<<<elemGrid, 256, 0, stream>>>(hbf, stateN);
    k_zero_bf<<<(CC + 255) / 256, 256, 0, stream>>>(zpad, CC);
    // xc = conv(x, k1) + b1 for all B*T at once
    k_conv_gemm<<<xGrid, gemmBlk, 0, stream>>>(abf, HWP * CC, zpad, wtA, b1,
                                               nullptr, 0, xc, HWP * N4, TB * TT * HWP);
    for (int t = 0; t < TT; t++) {
        k_conv_gemm<<<rGrid, gemmBlk, 0, stream>>>(hbf, HWP * FF, zpad, wtB, nullptr,
                                                   xc + (size_t)t * HWP * N4, TT * HWP * N4,
                                                   zb, HWP * N4, TB * HWP);
        k_lstm_gate<<<elemGrid, 256, 0, stream>>>(zb, cb, hbf, y1, t);
    }
    k_bn<<<bnGrid, 256, 0, stream>>>(y1, g1, be1, m1, v1, bnN);

    // -------- Layer 2 --------
    k_wconvert<<<wcvGrid, 256, 0, stream>>>(k2,  wtA);
    k_wconvert<<<wcvGrid, 256, 0, stream>>>(rk2, wtB);
    k_acvt<<<actGrid, 256, 0, stream>>>(y1, abf, actN);
    k_zero<<<elemGrid, 256, 0, stream>>>(cb, stateN);
    k_zero_bf<<<elemGrid, 256, 0, stream>>>(hbf, stateN);
    k_conv_gemm<<<xGrid, gemmBlk, 0, stream>>>(abf, HWP * FF, zpad, wtA, b2,
                                               nullptr, 0, xc, HWP * N4, TB * TT * HWP);
    for (int t = 0; t < TT; t++) {
        k_conv_gemm<<<rGrid, gemmBlk, 0, stream>>>(hbf, HWP * FF, zpad, wtB, nullptr,
                                                   xc + (size_t)t * HWP * N4, TT * HWP * N4,
                                                   zb, HWP * N4, TB * HWP);
        k_lstm_gate<<<elemGrid, 256, 0, stream>>>(zb, cb, hbf, y2, t);
    }

    // -------- BN2 + global average pool -> [B,T,1,1,512] --------
    const int poolN = TB * TT * FF;
    k_bnpool<<<(poolN + 255) / 256, 256, 0, stream>>>(y2, g2, be2, m2, v2, out);
}
